// MLPSequenceArithmetic_33062658244899
// MI455X (gfx1250) — compile-verified, matrix operand order checked by reading, not run on hardware
//
#include <hip/hip_runtime.h>

// CDNA5 / gfx1250: wave32, WMMA 16x16x32 f16 -> f32.
typedef __attribute__((ext_vector_type(16))) _Float16 v16h;
typedef __attribute__((ext_vector_type(8)))  float    v8f;
typedef __attribute__((ext_vector_type(8)))  _Float16 h8;
typedef __attribute__((ext_vector_type(2)))  _Float16 h2v;

#define SEQ_L 64
#define WPB   4     // waves per block
#define ROWS  16    // batch rows per wave (M of the WMMA tile)

static __device__ __forceinline__ v8f wmma_f16(v16h a, v16h b, v8f c) {
    // (neg_a, A, neg_b, B, c_mod, C, reuse_a, reuse_b)
    return __builtin_amdgcn_wmma_f32_16x16x32_f16(false, a, false, b, (short)0, c,
                                                  false, false);
}

// ReLU as a single v_max_num_f32: fast-math flags (nnan) scoped to this helper
// remove the backend's sNaN-canonicalize max.
static __device__ __forceinline__ float relu1(float x) {
    #pragma float_control(precise, off)
    return fmaxf(x, 0.0f);
}

// Two f32 -> packed f16 in one v_cvt_pk_rz_f16_f32.
static __device__ __forceinline__ h2v cvt_pk(float x, float y) {
    return __builtin_bit_cast(h2v, __builtin_amdgcn_cvt_pkrtz(x, y));
}

__global__ __launch_bounds__(WPB * 32)
void mlp_scan_kernel(const int* __restrict__ dgA, const int* __restrict__ dgB,
                     const float* __restrict__ Ea, const float* __restrict__ Eb,
                     const float* __restrict__ W1, const float* __restrict__ b1,
                     const float* __restrict__ W2, const float* __restrict__ b2,
                     const float* __restrict__ Wd, const float* __restrict__ bd,
                     const float* __restrict__ Wc, const float* __restrict__ bc,
                     float* __restrict__ outD, float* __restrict__ outC)
{
    // Per-wave staging: activations in f16 row-major [16 rows][64 cols],
    // carry logits in f32 [16 rows][2].
    __shared__ _Float16 hstage[WPB][16][64];
    __shared__ float    cstage[WPB][16][2];

    const int  lane = threadIdx.x & 31;
    const int  wave = threadIdx.x >> 5;
    const bool lolane = lane < 16;
    const int  n    = lane & 15;          // column id for B/C/D, row id for A
    const int  hi8  = lolane ? 0 : 8;     // D fragment row offset for hi half
    const int  m0   = (blockIdx.x * WPB + wave) * ROWS;

    // k index of element e in a 16-bit A/B WMMA fragment on this lane
    // (ISA 7.12.2: 16-bit A 16x32 layout).
    int kmap[16];
    #pragma unroll
    for (int e = 0; e < 16; ++e)
        kmap[e] = (e < 8) ? ((lolane ? 0 : 8) + e) : ((lolane ? 16 : 24) + (e - 8));

    // ---------------- Build weight B-fragments (resident in VGPRs) ----------
    // Layer 1: B[k][n] = W1[n][k], k<18 else 0; 4 tiles over the 64 outputs.
    v16h bw1[4];
    #pragma unroll
    for (int nt = 0; nt < 4; ++nt) {
        #pragma unroll
        for (int e = 0; e < 16; ++e) {
            const int k = kmap[e];
            const float w = (k < 18) ? W1[(nt * 16 + n) * 18 + k] : 0.0f;
            bw1[nt][e] = (_Float16)w;
        }
    }
    // Layer 2: K=64 split into two K=32 fragments; 2 N tiles over 32 outputs.
    v16h bw2[2][2];
    #pragma unroll
    for (int kt = 0; kt < 2; ++kt)
        #pragma unroll
        for (int nt = 0; nt < 2; ++nt)
            #pragma unroll
            for (int e = 0; e < 16; ++e)
                bw2[kt][nt][e] = (_Float16)W2[(nt * 16 + n) * 64 + (kt * 32 + kmap[e])];
    // Output layer: cols 0..9 = Wd, 10..11 = Wc, 12..15 = 0. K=32 exact.
    v16h bw3;
    #pragma unroll
    for (int e = 0; e < 16; ++e) {
        const int k = kmap[e];
        float w = 0.0f;
        if (n < 10)      w = Wd[n * 32 + k];
        else if (n < 12) w = Wc[(n - 10) * 32 + k];
        bw3[e] = (_Float16)w;
    }

    // Biases: one scalar per lane per tile (replicated into C rows at use).
    float bias1[4], bias2[2];
    #pragma unroll
    for (int nt = 0; nt < 4; ++nt) bias1[nt] = b1[nt * 16 + n];
    #pragma unroll
    for (int nt = 0; nt < 2; ++nt) bias2[nt] = b2[nt * 16 + n];
    const float bias3 = (n < 10) ? bd[n] : ((n < 12) ? bc[n - 10] : 0.0f);

    // Carry state (per batch row; duplicated on both half-waves).
    float c0 = 1.0f, c1 = 0.0f;

    const h8* hrow = (const h8*)&hstage[wave][n][0];  // 8x 16B chunks of row n

    #pragma unroll 1
    for (int t = 0; t < SEQ_L; ++t) {
        // ---- A1 fragment: lo half = Ea row + carry, hi half = Eb row -------
        const int dig = lolane ? dgA[(size_t)(m0 + n) * SEQ_L + t]
                               : dgB[(size_t)(m0 + n) * SEQ_L + t];
        const float4* e4 = (const float4*)((lolane ? Ea : Eb) + dig * 8);
        const float4 e4a = e4[0], e4b = e4[1];
        v16h a1;
        a1[0] = (_Float16)e4a.x; a1[1] = (_Float16)e4a.y;
        a1[2] = (_Float16)e4a.z; a1[3] = (_Float16)e4a.w;
        a1[4] = (_Float16)e4b.x; a1[5] = (_Float16)e4b.y;
        a1[6] = (_Float16)e4b.z; a1[7] = (_Float16)e4b.w;
        a1[8] = lolane ? (_Float16)c0 : (_Float16)0.0f;   // k=16: carry[0]
        a1[9] = lolane ? (_Float16)c1 : (_Float16)0.0f;   // k=17: carry[1]
        #pragma unroll
        for (int e = 10; e < 16; ++e) a1[e] = (_Float16)0.0f;

        // ---- Layer 1: h1 = relu(A1 @ W1^T + b1), staged to LDS as f16 ------
        #pragma unroll
        for (int nt = 0; nt < 4; ++nt) {
            v8f c;
            #pragma unroll
            for (int i = 0; i < 8; ++i) c[i] = bias1[nt];
            v8f d = wmma_f16(a1, bw1[nt], c);
            #pragma unroll
            for (int j = 0; j < 4; ++j) {  // rows 2j, 2j+1: one v_cvt_pk_rz
                const h2v p = cvt_pk(relu1(d[2 * j]), relu1(d[2 * j + 1]));
                hstage[wave][2 * j + hi8][nt * 16 + n]     = p[0];
                hstage[wave][2 * j + 1 + hi8][nt * 16 + n] = p[1];
            }
        }

        // ---- Re-fragment h1 as A operand (two K=32 fragments) --------------
        v16h a2[2];
        #pragma unroll
        for (int kt = 0; kt < 2; ++kt) {
            const h8 ch0 = hrow[kt * 4 + (lolane ? 0 : 1)];
            const h8 ch1 = hrow[kt * 4 + (lolane ? 2 : 3)];
            #pragma unroll
            for (int e = 0; e < 8; ++e) { a2[kt][e] = ch0[e]; a2[kt][8 + e] = ch1[e]; }
        }

        // ---- Layer 2: h2 = relu(h1 @ W2^T + b2), staged to LDS -------------
        #pragma unroll
        for (int nt = 0; nt < 2; ++nt) {
            v8f c;
            #pragma unroll
            for (int i = 0; i < 8; ++i) c[i] = bias2[nt];
            v8f d = wmma_f16(a2[0], bw2[0][nt], c);
            d = wmma_f16(a2[1], bw2[1][nt], d);
            #pragma unroll
            for (int j = 0; j < 4; ++j) {
                const h2v p = cvt_pk(relu1(d[2 * j]), relu1(d[2 * j + 1]));
                hstage[wave][2 * j + hi8][nt * 16 + n]     = p[0];
                hstage[wave][2 * j + 1 + hi8][nt * 16 + n] = p[1];
            }
        }

        // ---- Re-fragment h2 (single K=32 fragment) -------------------------
        v16h a3;
        {
            const h8 ch0 = hrow[lolane ? 0 : 1];
            const h8 ch1 = hrow[lolane ? 2 : 3];
            #pragma unroll
            for (int e = 0; e < 8; ++e) { a3[e] = ch0[e]; a3[8 + e] = ch1[e]; }
        }

        // ---- Output layer: cols 0..9 digit logits, 10..11 carry logits -----
        v8f c;
        #pragma unroll
        for (int i = 0; i < 8; ++i) c[i] = bias3;
        const v8f dO = wmma_f16(a3, bw3, c);

        #pragma unroll
        for (int r = 0; r < 8; ++r) {
            const int rowg = m0 + r + hi8;
            if (n < 10) {
                outD[(size_t)rowg * (SEQ_L * 10) + t * 10 + n] = dO[r];
            } else if (n < 12) {
                outC[(size_t)rowg * (SEQ_L * 2) + t * 2 + (n - 10)] = dO[r];
                cstage[wave][r + hi8][n - 10] = dO[r];   // broadcast via LDS
            }
        }

        // ---- softmax(carry_logit) -> next-step carry (per row n) -----------
        const float l0 = cstage[wave][n][0];
        const float l1 = cstage[wave][n][1];
        const float mx = fmaxf(l0, l1);
        const float x0 = __expf(l0 - mx);
        const float x1 = __expf(l1 - mx);
        const float inv = __builtin_amdgcn_rcpf(x0 + x1);  // v_rcp_f32 (TRANS)
        c0 = x0 * inv;
        c1 = x1 * inv;
    }
}

extern "C" void kernel_launch(void* const* d_in, const int* in_sizes, int n_in,
                              void* d_out, int out_size, void* d_ws, size_t ws_size,
                              hipStream_t stream) {
    const int*   A  = (const int*)d_in[0];
    const int*   B  = (const int*)d_in[1];
    const float* Ea = (const float*)d_in[2];
    const float* Eb = (const float*)d_in[3];
    const float* W1 = (const float*)d_in[4];
    const float* b1 = (const float*)d_in[5];
    const float* W2 = (const float*)d_in[6];
    const float* b2 = (const float*)d_in[7];
    const float* Wd = (const float*)d_in[8];
    const float* bd = (const float*)d_in[9];
    const float* Wc = (const float*)d_in[10];
    const float* bc = (const float*)d_in[11];

    float* out  = (float*)d_out;
    const int batch = in_sizes[0] / SEQ_L;          // 65536
    float* outD = out;                              // [B, L, 10]
    float* outC = out + (size_t)batch * SEQ_L * 10; // [B, L, 2]

    const int blocks = batch / (WPB * ROWS);        // 64 rows per block
    mlp_scan_kernel<<<blocks, WPB * 32, 0, stream>>>(
        A, B, Ea, Eb, W1, b1, W2, b2, Wd, bd, Wc, bc, outD, outC);
}
